// SigmoidAttention_68762426409365
// MI455X (gfx1250) — matrix operand order repaired and checked by reading, not compiled, not run on hardware
//
#include <hip/hip_runtime.h>
#include <hip/hip_bf16.h>

// B=512, T=128, C=768, NH=12, HD=64. Only q,k of the qkv projection are used.
// y[b] = (sum_h sigmoid_strictlower(q_h k_h^T / 8)) @ z[b] + 12*z[b]

typedef __attribute__((ext_vector_type(16))) __bf16 v16bf;
typedef __attribute__((ext_vector_type(8)))  float  v8f;

union FragBF { uint4 u[2]; v16bf v; };

__device__ __forceinline__ unsigned short f2bf(float f) {
    unsigned int u = __float_as_uint(f);
    u += 0x7fffu + ((u >> 16) & 1u);           // round-to-nearest-even
    return (unsigned short)(u >> 16);
}
// pack two f32 -> {lo: bf16(a), hi: bf16(b)} in one u32 (RNE)
__device__ __forceinline__ unsigned int f2bf_pk(float a, float b) {
    unsigned int ua = __float_as_uint(a), ub = __float_as_uint(b);
    ua += 0x7fffu + ((ua >> 16) & 1u);
    ub += 0x7fffu + ((ub >> 16) & 1u);
    return (ua >> 16) | (ub & 0xFFFF0000u);
}

// ---------------- TDM support (toolchain-dependent arity) -------------------
#if __has_include(<hip/amd_detail/amd_gfx1250_TDM.h>)
#define TDM_ARITY6 1
#else
#define TDM_ARITY6 0
#endif

#if __has_builtin(__builtin_amdgcn_tensor_load_to_lds) && \
    __has_builtin(__builtin_amdgcn_s_wait_tensorcnt)
#define HAVE_TDM 1
#else
#define HAVE_TDM 0
#endif

#if HAVE_TDM
typedef __attribute__((ext_vector_type(4))) unsigned int u32x4;
typedef __attribute__((ext_vector_type(8))) int i32x8;
typedef __attribute__((ext_vector_type(4))) int i32x4;

// DMA a [128 rows x 64 cols] bf16 tile (row stride 1536 elements) into LDS.
__device__ __forceinline__ void tdm_load_tile_128x64(const void* gsrc,
                                                     unsigned lds_off) {
    const unsigned long long ga = (unsigned long long)(size_t)gsrc;
    u32x4 g0;
    g0.x = 1u;                                   // count=1, user descriptor
    g0.y = lds_off;                              // lds_addr (bytes)
    g0.z = (unsigned)ga;                         // global_addr[31:0]
    g0.w = (unsigned)(ga >> 32) | (2u << 30);    // global_addr[56:32] | type=2
    i32x8 g1;
    g1[0] = 0x10000;                 // workgroup_mask=0, data_size=2B
    g1[1] = 0x06000000;              // tensor_dim0 = 1536 (lo16 at [63:48])
    g1[2] = 0x00800000;              // tensor_dim0 hi=0 | tensor_dim1=128 lo16
    g1[3] = 0x00400000;              // tensor_dim1 hi=0 | tile_dim0=64
    g1[4] = 128;                     // tile_dim1=128, tile_dim2=0
    g1[5] = 1536;                    // tensor_dim0_stride lo32
    g1[6] = 0;                       // stride hi | tensor_dim1_stride lo
    g1[7] = 0;
    i32x4 z4; z4[0] = z4[1] = z4[2] = z4[3] = 0;
#if TDM_ARITY6
    i32x8 z8; for (int i = 0; i < 8; ++i) z8[i] = 0;
    __builtin_amdgcn_tensor_load_to_lds(g0, g1, z4, z4, z8, 0);
#else
    __builtin_amdgcn_tensor_load_to_lds(g0, g1, z4, z4, 0);
#endif
}
#endif

// ---------------------------------------------------------------------------
// Kernel 1: qk[r, n] = sum_c x[r, c] * W[c, n] + bias[n],  r in [0,65536),
// n in [0,1536) (q columns then k columns; v never computed). Output bf16.
// Block tile: 128(M) x 64(N), K-steps of 32, 8 waves, 2x2 wmma tiles/wave.
// Grid: x = N (fast) so the 24 blocks sharing one x-tile are launch-adjacent
// and the tile stays in the 192 MB L2 (x streams from HBM once).
// ---------------------------------------------------------------------------
__global__ __launch_bounds__(256) void qk_proj_kernel(
    const float* __restrict__ x,        // [65536, 768]
    const float* __restrict__ W,        // [768, 2304]
    const float* __restrict__ bias,     // [2304]
    unsigned short* __restrict__ qk)    // [65536, 1536] bf16 bits
{
    __shared__ unsigned int   lds_x[128 * 16];   // [row][k/2] packed bf16 pairs, 8 KB
    __shared__ unsigned short lds_wt[64 * 32];   // [n][k]  4 KB (W transposed)

    const int tid  = threadIdx.x;
    const int lane = tid & 31;
    const int wave = tid >> 5;
    const int n0   = blockIdx.x * 64;    // 24 blocks in N (fast axis)
    const int m0   = blockIdx.y * 128;   // 512 blocks in M
    const int wm   = wave & 3;           // m subtiles 2*wm, 2*wm+1
    const int wn   = wave >> 2;          // n subtiles 2*wn, 2*wn+1
    const int lm   = lane & 15;
    const int hi   = lane >> 4;

    v8f acc[2][2];
    for (int i = 0; i < 2; ++i)
        for (int j = 0; j < 2; ++j)
            acc[i][j] = (v8f){0.f, 0.f, 0.f, 0.f, 0.f, 0.f, 0.f, 0.f};

    for (int k0 = 0; k0 < 768; k0 += 32) {
        // ---- stage x tile [128 x 32] f32 -> packed bf16 pairs in LDS ----
        #pragma unroll
        for (int pass = 0; pass < 4; ++pass) {
            int e4 = pass * 256 + tid;             // 1024 float4 total
            int r  = e4 >> 3;                      // 8 float4 per row
            int kk = (e4 & 7) * 4;
            const float4 f = *(const float4*)(x + (size_t)(m0 + r) * 768 + k0 + kk);
            uint2 p;
            p.x = f2bf_pk(f.x, f.y);
            p.y = f2bf_pk(f.z, f.w);
            *(uint2*)&lds_x[r * 16 + (kk >> 1)] = p;   // one ds_store_b64
        }
        // ---- stage W tile [32 x 64] -> transposed bf16 LDS [n][k] ----
        #pragma unroll
        for (int pass = 0; pass < 2; ++pass) {
            int e4 = pass * 256 + tid;             // 512 float4 total
            int kr = e4 >> 4;                      // 16 float4 per W row
            int nr = (e4 & 15) * 4;
            const float4 f = *(const float4*)(W + (size_t)(k0 + kr) * 2304 + n0 + nr);
            lds_wt[(nr + 0) * 32 + kr] = f2bf(f.x);
            lds_wt[(nr + 1) * 32 + kr] = f2bf(f.y);
            lds_wt[(nr + 2) * 32 + kr] = f2bf(f.z);
            lds_wt[(nr + 3) * 32 + kr] = f2bf(f.w);
        }
        // prefetch next x tile (gfx1250 global_prefetch_b8)
        if (k0 + 32 < 768)
            __builtin_prefetch(x + (size_t)(m0 + (tid >> 1)) * 768 + k0 + 32, 0, 1);
        __syncthreads();

        FragBF a[2], bw[2];
        #pragma unroll
        for (int i = 0; i < 2; ++i) {
            const unsigned short* p =
                (const unsigned short*)&lds_x[((2 * wm + i) * 16 + lm) * 16];
            a[i].u[0] = *(const uint4*)(p + 16 * hi);       // K = 16*hi .. 16*hi+7
            a[i].u[1] = *(const uint4*)(p + 16 * hi + 8);   // K = 16*hi+8 ..
        }
        #pragma unroll
        for (int j = 0; j < 2; ++j) {
            const unsigned short* p = lds_wt + ((2 * wn + j) * 16 + lm) * 32 + 16 * hi;
            bw[j].u[0] = *(const uint4*)(p);                // K = 16*hi .. +15
            bw[j].u[1] = *(const uint4*)(p + 8);
        }
        #pragma unroll
        for (int i = 0; i < 2; ++i)
            #pragma unroll
            for (int j = 0; j < 2; ++j)
                acc[i][j] = __builtin_amdgcn_wmma_f32_16x16x32_bf16(
                    false, a[i].v, false, bw[j].v, (short)0, acc[i][j], false, false);
        __syncthreads();
    }

    // ---- epilogue: + bias, store bf16 ----
    #pragma unroll
    for (int i = 0; i < 2; ++i) {
        #pragma unroll
        for (int j = 0; j < 2; ++j) {
            const int ncol = n0 + (2 * wn + j) * 16 + lm;
            const float bv = bias[ncol];
            #pragma unroll
            for (int r = 0; r < 8; ++r) {
                const int row = m0 + (2 * wm + i) * 16 + r + 8 * hi;
                qk[(size_t)row * 1536 + ncol] = f2bf(acc[i][j][r] + bv);
            }
        }
    }
}

// ---------------------------------------------------------------------------
// Kernel 2: one block per batch b. Phase 1: A = sum_h sigmoid(q_h k_h^T / 8)
// on the strict-lower triangle. q_h/k_h tiles are DMA'd into double-buffered
// LDS stages by the Tensor Data Mover, overlapping DMA of head h+1 with the
// WMMA+sigmoid of head h (fallback: direct per-lane global loads). Phase 2:
// y[b] = A @ z[b] + 12*z[b], with z staged transposed in LDS per 128-col chunk.
//
// LDS element map (unsigned short units):
//   [0, 16384)          A_bf [128][128]
//   [16384, 24576)      stage q buf0   [128][64]
//   [24576, 32768)      stage k buf0   [128][64]
//   [32768, 40960)      stage q buf1   [128][64]
//   [40960, 49152)      stage k buf1   [128][64]
//   [16384, 32768)      zT [128][128]  (phase 2, reuses buf0 region)
// ---------------------------------------------------------------------------
#define A_ELEMS   (128 * 128)
#define STG_ELEMS (128 * 64)

__global__ __launch_bounds__(256) void attn_combine_kernel(
    const unsigned short* __restrict__ qk,  // [65536, 1536] bf16
    const float* __restrict__ z,            // [512, 128, 768]
    float* __restrict__ y)                  // [512, 128, 768]
{
    extern __shared__ unsigned short smem[];
    unsigned short* A_bf = smem;                  // 32 KB
    unsigned short* zT   = smem + A_ELEMS;        // 32 KB (phase 2)

    const int b    = blockIdx.x;
    const int tid  = threadIdx.x;
    const int lane = tid & 31;
    const int wave = tid >> 5;        // wave == t-tile index (rows 16w..16w+15)
    const int lm   = lane & 15;
    const int hi   = lane >> 4;

    const unsigned short* qb = qk + (size_t)b * 128 * 1536;       // q at col 0
    const unsigned short* kb = qb + 768;                          // k at col 768
    const float* zb = z + (size_t)b * 128 * 768;
    float*       yb = y + (size_t)b * 128 * 768;

    // ---------------- Phase 1: head-summed masked sigmoid scores -----------
    v8f acc[8];
    #pragma unroll
    for (int ts = 0; ts < 8; ++ts)
        acc[ts] = (v8f){0.f, 0.f, 0.f, 0.f, 0.f, 0.f, 0.f, 0.f};

#if HAVE_TDM
    if (wave == 0) {   // prime the pipeline: head 0 into buffer 0
        unsigned short* q0 = smem + A_ELEMS;
        tdm_load_tile_128x64(qb, (unsigned)(size_t)(void*)q0);
        tdm_load_tile_128x64(kb, (unsigned)(size_t)(void*)(q0 + STG_ELEMS));
    }
#endif
    for (int h = 0; h < 12; ++h) {
        const int buf = h & 1;
        unsigned short* qsb  = smem + A_ELEMS + buf * (2 * STG_ELEMS);
        unsigned short* kstb = qsb + STG_ELEMS;
#if HAVE_TDM
        if (wave == 0) {
            if (h + 1 < 12) {   // overlap next head's DMA with this compute
                unsigned short* qn = smem + A_ELEMS + (buf ^ 1) * (2 * STG_ELEMS);
                tdm_load_tile_128x64(qb + (h + 1) * 64,
                                     (unsigned)(size_t)(void*)qn);
                tdm_load_tile_128x64(kb + (h + 1) * 64,
                                     (unsigned)(size_t)(void*)(qn + STG_ELEMS));
                __builtin_amdgcn_s_wait_tensorcnt((short)2);  // head h landed
            } else {
                __builtin_amdgcn_s_wait_tensorcnt((short)0);
            }
        }
        __syncthreads();
#endif
        FragBF aq[2];
        #pragma unroll
        for (int ks = 0; ks < 2; ++ks) {
#if HAVE_TDM
            const unsigned short* p = qsb + (wave * 16 + lm) * 64 + ks * 32;
#else
            const unsigned short* p =
                qb + (size_t)(wave * 16 + lm) * 1536 + h * 64 + ks * 32;
#endif
            aq[ks].u[0] = *(const uint4*)(p + 16 * hi);
            aq[ks].u[1] = *(const uint4*)(p + 16 * hi + 8);
        }
        #pragma unroll
        for (int ts = 0; ts < 8; ++ts) {
            if (ts > wave) continue;   // tiles entirely above diagonal are zero
            FragBF bk[2];
            #pragma unroll
            for (int ks = 0; ks < 2; ++ks) {
#if HAVE_TDM
                const unsigned short* p =
                    kstb + (ts * 16 + lm) * 64 + ks * 32 + 16 * hi;
#else
                const unsigned short* p =
                    kb + (size_t)(ts * 16 + lm) * 1536 + h * 64 + ks * 32 + 16 * hi;
#endif
                bk[ks].u[0] = *(const uint4*)(p);
                bk[ks].u[1] = *(const uint4*)(p + 8);
            }
            v8f s = (v8f){0.f, 0.f, 0.f, 0.f, 0.f, 0.f, 0.f, 0.f};
            s = __builtin_amdgcn_wmma_f32_16x16x32_bf16(
                    false, aq[0].v, false, bk[0].v, (short)0, s, false, false);
            s = __builtin_amdgcn_wmma_f32_16x16x32_bf16(
                    false, aq[1].v, false, bk[1].v, (short)0, s, false, false);
            const int ss = ts * 16 + lm;             // column (key index)
            #pragma unroll
            for (int r = 0; r < 8; ++r) {
                const int tt = wave * 16 + r + 8 * hi;   // row (query index)
                const float sg =
                    __builtin_amdgcn_rcpf(1.0f + __expf(-s[r] * 0.125f));
                acc[ts][r] += (ss < tt) ? sg : 0.0f;
            }
        }
#if HAVE_TDM
        __syncthreads();   // all waves done with buf before it is re-DMA'd
#endif
    }
    // dump strip to LDS as bf16 (upper tiles written as exact zero)
    #pragma unroll
    for (int ts = 0; ts < 8; ++ts) {
        #pragma unroll
        for (int r = 0; r < 8; ++r) {
            const int tt = wave * 16 + r + 8 * hi;
            const int ss = ts * 16 + lm;
            A_bf[tt * 128 + ss] = (ts <= wave) ? f2bf(acc[ts][r]) : (unsigned short)0;
        }
    }
    __syncthreads();

    // ---------------- Phase 2: y = A @ z + 12*z, 128-column chunks ---------
    for (int chunk = 0; chunk < 6; ++chunk) {
        const int d0 = chunk * 128;
        __syncthreads();   // previous chunk's consumers done before overwrite
        for (int pass = 0; pass < 16; ++pass) {
            int e4 = pass * 256 + tid;            // 4096 float4 total
            int s  = e4 >> 5;                     // 32 float4 per z row-chunk
            int dd = (e4 & 31) * 4;
            const float4 f = *(const float4*)(zb + (size_t)s * 768 + d0 + dd);
            zT[(dd + 0) * 128 + s] = f2bf(f.x);
            zT[(dd + 1) * 128 + s] = f2bf(f.y);
            zT[(dd + 2) * 128 + s] = f2bf(f.z);
            zT[(dd + 3) * 128 + s] = f2bf(f.w);
        }
        __syncthreads();

        #pragma unroll
        for (int nt = 0; nt < 8; ++nt) {
            v8f yacc = (v8f){0.f, 0.f, 0.f, 0.f, 0.f, 0.f, 0.f, 0.f};
            #pragma unroll
            for (int ks = 0; ks < 4; ++ks) {
                FragBF a, bz;
                const unsigned short* pa = A_bf + (wave * 16 + lm) * 128 + ks * 32;
                a.u[0] = *(const uint4*)(pa + 16 * hi);
                a.u[1] = *(const uint4*)(pa + 16 * hi + 8);
                const unsigned short* pb = zT + (nt * 16 + lm) * 128 + ks * 32 + 16 * hi;
                bz.u[0] = *(const uint4*)(pb);
                bz.u[1] = *(const uint4*)(pb + 8);
                yacc = __builtin_amdgcn_wmma_f32_16x16x32_bf16(
                           false, a.v, false, bz.v, (short)0, yacc, false, false);
            }
            #pragma unroll
            for (int r = 0; r < 8; ++r) {
                const int tt = wave * 16 + r + 8 * hi;
                const int dd = d0 + nt * 16 + lm;
                yb[(size_t)tt * 768 + dd] = yacc[r] + 12.0f * zb[(size_t)tt * 768 + dd];
            }
        }
    }
}

// ---------------------------------------------------------------------------
extern "C" void kernel_launch(void* const* d_in, const int* in_sizes, int n_in,
                              void* d_out, int out_size, void* d_ws, size_t ws_size,
                              hipStream_t stream) {
    const float* x    = (const float*)d_in[0];   // [512,128,768]
    const float* z    = (const float*)d_in[1];   // [512,128,768]
    const float* W    = (const float*)d_in[2];   // [768,2304]
    const float* bias = (const float*)d_in[3];   // [2304]
    float* y = (float*)d_out;                    // [512,128,768]

    unsigned short* qk = (unsigned short*)d_ws;  // 65536*1536 bf16 = 201 MB

    dim3 g1(24, 512), blk(256);                  // N fast => x-tile L2 reuse
    qk_proj_kernel<<<g1, blk, 0, stream>>>(x, W, bias, qk);

    // 32 KB A + 2 x (16 KB q + 16 KB k) double-buffered TDM stages = 96 KB
    const size_t lds2 = (size_t)(A_ELEMS + 4 * STG_ELEMS) * sizeof(unsigned short);
    attn_combine_kernel<<<dim3(512), blk, lds2, stream>>>(qk, z, y);
}
